// stGuide_Integration_Model_85272280695562
// MI455X (gfx1250) — matrix operand order repaired and checked by reading, not compiled, not run on hardware
//
#include <hip/hip_runtime.h>
#include <hip/hip_bf16.h>

// ---------------------------------------------------------------------------
// Model constants (match reference)
// ---------------------------------------------------------------------------
constexpr int NODES = 20000, EDGES = 120000;
constexpr int F_IN = 3000, F_HID = 512, F_OUT = 30;
constexpr int HEADS = 3, NDOMS = 4, MHID = 64, NLABEL = 20;
constexpr float EPS_BN = 1e-5f;

// Padded GEMM geometry (all tile-multiple, zero-filled in workspace)
constexpr int MP    = 20096;  // 157 * 128   (rows, padded)
constexpr int KP_IN = 3008;   // 94 * 32     (K for F_IN GEMMs)
constexpr int NP_IN = 3072;   // 24 * 128    (N for F_IN outputs)
constexpr int NP_H1 = 1536;   // 12 * 128
constexpr int KP_H  = 512;    // 16 * 32
constexpr int NP_O  = 128;    // 1 * 128     (HEADS*F_OUT=90 padded)

#define USE_TDM 1   // Tensor Data Mover staging; set 0 to fall back to vector loads

typedef __attribute__((ext_vector_type(16))) __bf16 v16bf;
typedef __attribute__((ext_vector_type(8)))  float  v8f;
typedef unsigned int u32x4 __attribute__((ext_vector_type(4)));
typedef int          i32x8 __attribute__((ext_vector_type(8)));

struct U32x8 { uint4 lo, hi; };   // 32 bytes, bit-cast target for v16bf

__device__ __forceinline__ float eluf(float x) { return x > 0.f ? x : __expf(x) - 1.f; }

__device__ __forceinline__ void atomicMaxFloat(float* addr, float val) {
  if (val >= 0.f) atomicMax((int*)addr, __float_as_int(val));
  else            atomicMin((unsigned int*)addr, (unsigned int)__float_as_int(val));
}

// ---------------------------------------------------------------------------
// Pack f32 [rows x cols] into zero-padded bf16 [prows x ldo]
// ---------------------------------------------------------------------------
__global__ void cast_pad_kernel(const float* __restrict__ in, __bf16* __restrict__ out,
                                int rows, int cols, int prows, int ldo) {
  size_t i = blockIdx.x * (size_t)blockDim.x + threadIdx.x;
  if (i >= (size_t)prows * ldo) return;
  int r = (int)(i / ldo), c = (int)(i % ldo);
  float v = (r < rows && c < cols) ? in[(size_t)r * cols + c] : 0.f;
  out[i] = (__bf16)v;
}

// Transpose f32 B[K x N] into zero-padded K-major bf16 Bt[np x ldo] (row n holds K)
__global__ void transpose_pad_kernel(const float* __restrict__ in, __bf16* __restrict__ out,
                                     int K, int N, int np, int ldo) {
  size_t i = blockIdx.x * (size_t)blockDim.x + threadIdx.x;
  if (i >= (size_t)np * ldo) return;
  int n = (int)(i / ldo), k = (int)(i % ldo);
  float v = (k < K && n < N) ? in[(size_t)k * N + n] : 0.f;
  out[i] = (__bf16)v;
}

__global__ void fill_f32_kernel(float* __restrict__ p, float v, size_t n) {
  size_t i = blockIdx.x * (size_t)blockDim.x + threadIdx.x;
  if (i < n) p[i] = v;
}

// ---------------------------------------------------------------------------
// WMMA bf16 GEMM:  C[M,N] = A[M,K] * Bt[N,K]^T (+bias), f32 accumulate
//   A  : [>=gridY*128 x lda] bf16, K-contiguous, fully padded (no guards)
//   Bt : [>=gridX*128 x ldb] bf16, K-contiguous, fully padded (no guards)
//   C  : compact [M x N] f32, guarded store
// Block tile 128x128x32, 8 waves as 4(M) x 2(N); wave tile 32x64 -> 8 WMMAs.
// Double-buffered LDS; tiles staged by the Tensor Data Mover (one 2D D# per
// operand), DMA of tile kt+1 overlapped with WMMA compute on tile kt.
// ---------------------------------------------------------------------------
__global__ __launch_bounds__(256)
void gemm_bf16_wmma(const __bf16* __restrict__ A, const __bf16* __restrict__ Bt,
                    float* __restrict__ C, const float* __restrict__ bias,
                    int M, int N, int K, int lda, int ldb)
{
  __shared__ __bf16 As[2][128][32];   // [buf][m][k]
  __shared__ __bf16 Bs[2][128][32];   // [buf][n][k]

  const int tid  = threadIdx.x;
  const int lane = tid & 31;
  const int wave = tid >> 5;
  const int bm = blockIdx.y * 128;
  const int bn = blockIdx.x * 128;

  const int wr = (wave & 3) * 32;   // wave row offset within tile
  const int wc = (wave >> 2) * 64;  // wave col offset within tile

  v8f acc[2][4] = {};

  const int KT = (K + 31) / 32;     // number of k-tiles (padded data covers all)

#if USE_TDM
  // Issue one 128x32 bf16 tile DMA (wave0 -> A, wave1 -> B) into LDS buffer `buf`.
  auto issue_tile = [&](int k0, int buf) {
    if (wave < 2) {
      const bool isA = (wave == 0);
      const __bf16* gsrc = isA ? (A + (size_t)bm * lda + k0)
                               : (Bt + (size_t)bn * ldb + k0);
      const unsigned lds_addr = (unsigned)(size_t)(isA ? &As[buf][0][0] : &Bs[buf][0][0]);
      const unsigned long long ga = (unsigned long long)(size_t)gsrc;
      const unsigned stride = (unsigned)(isA ? lda : ldb);

      u32x4 g0;
      g0[0] = 1u;                                   // count=1, user descriptor
      g0[1] = lds_addr;                             // lds_addr[31:0]
      g0[2] = (unsigned)ga;                         // global_addr[31:0]
      g0[3] = (unsigned)(ga >> 32) | (2u << 30);    // global_addr[56:32] | type=2

      i32x8 g1;
      g1[0] = (1 << 16);          // workgroup_mask=0, data_size=1 (2 bytes)
      g1[1] = (32 << 16);         // tensor_dim0[15:0]=32 in [31:16]
      g1[2] = (128 << 16);        // tensor_dim0 hi=0 | tensor_dim1[15:0]=128
      g1[3] = (32 << 16);         // tensor_dim1 hi=0 | tile_dim0=32
      g1[4] = 128;                // tile_dim1=128 | tile_dim2=0
      g1[5] = (int)stride;        // tensor_dim0_stride[31:0] (elements)
      g1[6] = 0;                  // stride hi | tensor_dim1_stride lo (unused, 2D)
      g1[7] = 0;

      asm volatile("tensor_load_to_lds %0, %1" :: "s"(g0), "s"(g1) : "memory");
    }
  };

  // Prologue: fill buffer 0 with tile 0.
  issue_tile(0, 0);
  if (wave < 2) __builtin_amdgcn_s_wait_tensorcnt(0);
  __syncthreads();
#else
  // Fallback prologue: cooperative 16B vector staging into buffer 0.
  #pragma unroll
  for (int c = tid; c < 512; c += 256) {
    int row = c >> 2, ks = (c & 3) * 8;
    *(uint4*)&As[0][row][ks] = *(const uint4*)(A + (size_t)(bm + row) * lda + ks);
    *(uint4*)&Bs[0][row][ks] = *(const uint4*)(Bt + (size_t)(bn + row) * ldb + ks);
  }
  __syncthreads();
#endif

  for (int kt = 0; kt < KT; ++kt) {
    const int cur = kt & 1;
    const bool more = (kt + 1) < KT;

#if USE_TDM
    // Kick off DMA of the NEXT tile into the other buffer; overlaps compute below.
    if (more) issue_tile((kt + 1) * 32, cur ^ 1);
#else
    if (more) {
      #pragma unroll
      for (int c = tid; c < 512; c += 256) {
        int row = c >> 2, ks = (c & 3) * 8;
        *(uint4*)&As[cur ^ 1][row][ks] =
            *(const uint4*)(A + (size_t)(bm + row) * lda + (kt + 1) * 32 + ks);
        *(uint4*)&Bs[cur ^ 1][row][ks] =
            *(const uint4*)(Bt + (size_t)(bn + row) * ldb + (kt + 1) * 32 + ks);
      }
    }
#endif

    // ---- build fragments (ISA 16-bit A/B VGPR layouts) ----
    // A 16x32: lanes 0-15 -> M=lane, K {0..7,16..23}; lanes 16-31 -> M=lane-16, K {8..15,24..31}
    const int arow = wr + (lane & 15);
    const int akb  = (lane < 16) ? 0 : 8;
    v16bf af[2];
#pragma unroll
    for (int mi = 0; mi < 2; ++mi) {
      U32x8 t;
      t.lo = *(const uint4*)&As[cur][arow + mi * 16][akb];
      t.hi = *(const uint4*)&As[cur][arow + mi * 16][akb + 16];
      af[mi] = __builtin_bit_cast(v16bf, t);
    }
    // B 32x16: lanes 0-15 -> N=lane, K {0..7,16..23}; lanes 16-31 -> N=lane-16, K {8..15,24..31}
    const int bkb = (lane < 16) ? 0 : 8;
    v16bf bfr[4];
#pragma unroll
    for (int ni = 0; ni < 4; ++ni) {
      const int brow = wc + ni * 16 + (lane & 15);
      U32x8 t;
      t.lo = *(const uint4*)&Bs[cur][brow][bkb];
      t.hi = *(const uint4*)&Bs[cur][brow][bkb + 16];
      bfr[ni] = __builtin_bit_cast(v16bf, t);
    }

#pragma unroll
    for (int mi = 0; mi < 2; ++mi)
#pragma unroll
      for (int ni = 0; ni < 4; ++ni)
        acc[mi][ni] = __builtin_amdgcn_wmma_f32_16x16x32_bf16(
            false, af[mi], false, bfr[ni], (short)0, acc[mi][ni], false, false);

#if USE_TDM
    // Issuing waves drain their tensor counter, then everyone syncs: this both
    // publishes the freshly DMA'd buffer and retires all reads of `cur` before
    // the next iteration's DMA may overwrite it.
    if (more && wave < 2) __builtin_amdgcn_s_wait_tensorcnt(0);
#endif
    __syncthreads();
  }

  // ---- store C (f32 C/D layout: VGPR r -> M=r / 8+r, N=lane / lane-16) ----
  const int cm = bm + wr + ((lane < 16) ? 0 : 8);
  const int cn = bn + wc + (lane & 15);
#pragma unroll
  for (int mi = 0; mi < 2; ++mi) {
#pragma unroll
    for (int ni = 0; ni < 4; ++ni) {
      const int col = cn + ni * 16;
      if (col < N) {
        const float bv = bias ? bias[col] : 0.f;
#pragma unroll
        for (int r = 0; r < 8; ++r) {
          const int row = cm + mi * 16 + r;
          if (row < M) C[(size_t)row * N + col] = acc[mi][ni][r] + bv;
        }
      }
    }
  }
}

// ---------------------------------------------------------------------------
// BatchNorm: column partial sums (coalesced: thread per column, chunk of rows)
// ---------------------------------------------------------------------------
__global__ void colsum_partial(const float* __restrict__ x, float* __restrict__ sum,
                               float* __restrict__ sq, int rows, int cols, int chunk) {
  int c = blockIdx.x * blockDim.x + threadIdx.x;
  if (c >= cols) return;
  int r0 = blockIdx.y * chunk;
  int r1 = r0 + chunk; if (r1 > rows) r1 = rows;
  float s = 0.f, s2 = 0.f;
  for (int r = r0; r < r1; ++r) { float v = x[(size_t)r * cols + c]; s += v; s2 += v * v; }
  atomicAdd(&sum[c], s);
  atomicAdd(&sq[c], s2);
}

__global__ void bn_finalize(const float* __restrict__ sum, const float* __restrict__ sq,
                            float* __restrict__ mean, float* __restrict__ var,
                            int rows, int cols) {
  int c = blockIdx.x * blockDim.x + threadIdx.x;
  if (c >= cols) return;
  float m = sum[c] / (float)rows;
  float v = sq[c] / (float)rows - m * m;
  mean[c] = m;
  var[c] = v > 0.f ? v : 0.f;
}

// normalize+ELU, writing zero-padded bf16 [MP x ldo]
__global__ void bn_apply_elu_kernel(const float* __restrict__ x, const float* __restrict__ mean,
                                    const float* __restrict__ var, const float* __restrict__ g,
                                    const float* __restrict__ b, __bf16* __restrict__ out,
                                    int rows, int cols, int prows, int ldo) {
  size_t i = blockIdx.x * (size_t)blockDim.x + threadIdx.x;
  if (i >= (size_t)prows * ldo) return;
  int r = (int)(i / ldo), c = (int)(i % ldo);
  float res = 0.f;
  if (r < rows && c < cols) {
    float xv = x[(size_t)r * cols + c];
    float xn = (xv - mean[c]) * rsqrtf(var[c] + EPS_BN) * g[c] + b[c];
    res = eluf(xn);
  }
  out[i] = (__bf16)res;
}

// ---------------------------------------------------------------------------
// GAT attention machinery
// ---------------------------------------------------------------------------
__global__ void gat_logits(const float* __restrict__ hproj, const float* __restrict__ a_s,
                           const float* __restrict__ a_d, float* __restrict__ al_s,
                           float* __restrict__ al_d, int nodes, int F) {
  int idx = blockIdx.x * blockDim.x + threadIdx.x;
  if (idx >= nodes * HEADS) return;
  int n = idx / HEADS, h = idx % HEADS;
  const float* hp = hproj + (size_t)n * HEADS * F + (size_t)h * F;
  const float* as = a_s + (size_t)h * F;
  const float* ad = a_d + (size_t)h * F;
  float ss = 0.f, sd = 0.f;
  for (int d = 0; d < F; ++d) { float v = hp[d]; ss += v * as[d]; sd += v * ad[d]; }
  al_s[idx] = ss;
  al_d[idx] = sd;
}

__global__ void edge_max_kernel(const int* __restrict__ src, const int* __restrict__ dst,
                                const float* __restrict__ al_s, const float* __restrict__ al_d,
                                float* __restrict__ e, float* __restrict__ m) {
  int idx = blockIdx.x * blockDim.x + threadIdx.x;
  if (idx >= EDGES * HEADS) return;
  int ed = idx / HEADS, h = idx % HEADS;
  int s = src[ed], d = dst[ed];
  float v = al_s[s * HEADS + h] + al_d[d * HEADS + h];
  v = v > 0.f ? v : 0.2f * v;   // leaky_relu(0.2)
  e[idx] = v;
  atomicMaxFloat(&m[d * HEADS + h], v);
}

__global__ void edge_exp_sum_kernel(const int* __restrict__ dst, float* __restrict__ e,
                                    const float* __restrict__ m, float* __restrict__ den) {
  int idx = blockIdx.x * blockDim.x + threadIdx.x;
  if (idx >= EDGES * HEADS) return;
  int ed = idx / HEADS, h = idx % HEADS;
  int d = dst[ed];
  float ex = __expf(e[idx] - m[d * HEADS + h]);
  e[idx] = ex;
  atomicAdd(&den[d * HEADS + h], ex);
}

__global__ void scatter_msg_kernel(const int* __restrict__ src, const int* __restrict__ dst,
                                   const float* __restrict__ e, const float* __restrict__ den,
                                   const float* __restrict__ hproj, float* __restrict__ out,
                                   int F) {
  size_t idx = blockIdx.x * (size_t)blockDim.x + threadIdx.x;
  size_t total = (size_t)EDGES * HEADS * F;
  if (idx >= total) return;
  int d = (int)(idx % F);
  size_t t = idx / F;
  int h = (int)(t % HEADS);
  int ed = (int)(t / HEADS);
  int s = src[ed], dd = dst[ed];
  float coef = e[(size_t)ed * HEADS + h] / (den[dd * HEADS + h] + 1e-16f);
  atomicAdd(&out[(size_t)dd * HEADS * F + (size_t)h * F + d],
            hproj[(size_t)s * HEADS * F + (size_t)h * F + d] * coef);
}

__global__ void head_mean_kernel(const float* __restrict__ in, float* __restrict__ of,
                                 __bf16* __restrict__ ob, int nodes, int F, int do_elu) {
  size_t i = blockIdx.x * (size_t)blockDim.x + threadIdx.x;
  if (i >= (size_t)nodes * F) return;
  int n = (int)(i / F), d = (int)(i % F);
  const float* p = in + (size_t)n * HEADS * F + d;
  float s = 0.f;
#pragma unroll
  for (int h = 0; h < HEADS; ++h) s += p[(size_t)h * F];
  s *= (1.f / HEADS);
  if (do_elu) s = eluf(s);
  if (of) of[i] = s;
  if (ob) ob[i] = (__bf16)s;
}

// ---------------------------------------------------------------------------
// MLP head (per-node, tiny)
// ---------------------------------------------------------------------------
__global__ void mlp_kernel(const float* __restrict__ z, const float* __restrict__ w1,
                           const float* __restrict__ b1, const float* __restrict__ w2,
                           const float* __restrict__ b2, const int* __restrict__ flag,
                           float* __restrict__ yp) {
  int n = blockIdx.x * blockDim.x + threadIdx.x;
  if (n >= NODES) return;
  if (*flag == 0) {
    for (int j = 0; j < NLABEL; ++j) yp[(size_t)n * NLABEL + j] = 0.f;
    return;
  }
  float zr[F_OUT];
  for (int k = 0; k < F_OUT; ++k) zr[k] = z[(size_t)n * F_OUT + k];
  float hid[MHID];
  for (int j = 0; j < MHID; ++j) {
    float s = b1[j];
    for (int k = 0; k < F_OUT; ++k) s += zr[k] * w1[k * MHID + j];
    hid[j] = s > 0.f ? s : 0.f;
  }
  for (int j = 0; j < NLABEL; ++j) {
    float s = b2[j];
    for (int k = 0; k < MHID; ++k) s += hid[k] * w2[k * NLABEL + j];
    yp[(size_t)n * NLABEL + j] = s;
  }
}

// ---------------------------------------------------------------------------
// fc2: [N,30] @ [30,3000] + b  (K tiny -> plain FMA, coalesced over columns)
// ---------------------------------------------------------------------------
__global__ void fc2_kernel(const float* __restrict__ z, const float* __restrict__ w,
                           const float* __restrict__ b, float* __restrict__ out) {
  size_t i = blockIdx.x * (size_t)blockDim.x + threadIdx.x;
  if (i >= (size_t)NODES * F_IN) return;
  int n = (int)(i / F_IN), c = (int)(i % F_IN);
  float s = b[c];
  const float* zp = z + (size_t)n * F_OUT;
#pragma unroll 6
  for (int k = 0; k < F_OUT; ++k) s += zp[k] * w[(size_t)k * F_IN + c];
  out[i] = s;
}

// ---------------------------------------------------------------------------
// Domain-specific BN
// ---------------------------------------------------------------------------
__global__ void dsbn_count_kernel(const int* __restrict__ y, int* __restrict__ cnt) {
  int n = blockIdx.x * blockDim.x + threadIdx.x;
  if (n < NODES) atomicAdd(&cnt[y[n]], 1);
}

__global__ void dsbn_partial(const float* __restrict__ x, const int* __restrict__ y,
                             float* __restrict__ sum, float* __restrict__ sq,
                             int rows, int cols, int chunk) {
  int c = blockIdx.x * blockDim.x + threadIdx.x;
  if (c >= cols) return;
  int r0 = blockIdx.y * chunk;
  int r1 = r0 + chunk; if (r1 > rows) r1 = rows;
  float s[NDOMS] = {}, s2[NDOMS] = {};
  for (int r = r0; r < r1; ++r) {
    float v = x[(size_t)r * cols + c];
    int d = y[r];
#pragma unroll
    for (int dd = 0; dd < NDOMS; ++dd)
      if (d == dd) { s[dd] += v; s2[dd] += v * v; }
  }
#pragma unroll
  for (int d = 0; d < NDOMS; ++d) {
    atomicAdd(&sum[d * cols + c], s[d]);
    atomicAdd(&sq[d * cols + c], s2[d]);
  }
}

// DSBN apply + ELU, writing zero-padded bf16 [MP x ldo]
__global__ void dsbn_apply_kernel(const float* __restrict__ x, const int* __restrict__ y,
                                  const int* __restrict__ cnt, const float* __restrict__ sum,
                                  const float* __restrict__ sq, const float* __restrict__ g,
                                  const float* __restrict__ b, __bf16* __restrict__ out,
                                  int rows, int cols, int prows, int ldo) {
  size_t i = blockIdx.x * (size_t)blockDim.x + threadIdx.x;
  if (i >= (size_t)prows * ldo) return;
  int r = (int)(i / ldo), c = (int)(i % ldo);
  float res = 0.f;
  if (r < rows && c < cols) {
    int d = y[r];
    float cf = (float)cnt[d];
    float safe = cf > 1.f ? cf : 1.f;
    float m = sum[d * cols + c] / safe;
    float v = sq[d * cols + c] / safe - m * m;
    v = v > 0.f ? v : 0.f;
    float xv = x[(size_t)r * cols + c];
    float xn = (xv - m) * rsqrtf(v + EPS_BN) * g[d * cols + c] + b[d * cols + c];
    res = eluf((cf > 1.f) ? xn : xv);
  }
  out[i] = (__bf16)res;
}

// ---------------------------------------------------------------------------
// Host orchestration
// ---------------------------------------------------------------------------
static inline unsigned grid1(size_t n, int blk) { return (unsigned)((n + blk - 1) / blk); }

extern "C" void kernel_launch(void* const* d_in, const int* in_sizes, int n_in,
                              void* d_out, int out_size, void* d_ws, size_t ws_size,
                              hipStream_t stream) {
  (void)in_sizes; (void)n_in; (void)out_size; (void)ws_size;

  const float* features = (const float*)d_in[0];
  const int*   edge     = (const int*)d_in[1];
  const int*   y        = (const int*)d_in[2];
  const int*   mlp_flag = (const int*)d_in[3];
  const float* fc1_w = (const float*)d_in[4];
  const float* fc1_b = (const float*)d_in[5];
  const float* bn_g  = (const float*)d_in[6];
  const float* bn_b  = (const float*)d_in[7];
  const float* W1    = (const float*)d_in[8];
  const float* a1_s  = (const float*)d_in[9];
  const float* a1_d  = (const float*)d_in[10];
  const float* W2    = (const float*)d_in[11];
  const float* a2_s  = (const float*)d_in[12];
  const float* a2_d  = (const float*)d_in[13];
  const float* mw1   = (const float*)d_in[14];
  const float* mb1   = (const float*)d_in[15];
  const float* mw2   = (const float*)d_in[16];
  const float* mb2   = (const float*)d_in[17];
  const float* fc2_w = (const float*)d_in[18];
  const float* fc2_b = (const float*)d_in[19];
  const float* dg    = (const float*)d_in[20];
  const float* db    = (const float*)d_in[21];
  const float* fc3_w = (const float*)d_in[22];
  const float* fc3_b = (const float*)d_in[23];

  const int* src = edge;
  const int* dst = edge + EDGES;

  float* out_z  = (float*)d_out;                   // [N,30]
  float* out_h3 = out_z + (size_t)NODES * F_OUT;   // [N,3000]
  float* out_yp = out_h3 + (size_t)NODES * F_IN;   // [N,20]

  // ---- workspace carving ----
  char* p = (char*)d_ws;
  auto carve = [&](size_t bytes) -> char* {
    char* r = p; p += (bytes + 255) & ~(size_t)255; return r;
  };
  __bf16* wb_fc1 = (__bf16*)carve((size_t)NP_IN * KP_IN * 2);   // fc1_w^T, padded
  __bf16* wb_w1  = (__bf16*)carve((size_t)NP_H1 * KP_IN * 2);   // W1^T, padded
  __bf16* wb_w2  = (__bf16*)carve((size_t)NP_O * KP_H * 2);     // W2^T, padded
  __bf16* wb_fc3 = (__bf16*)carve((size_t)NP_IN * KP_IN * 2);   // fc3_w^T, padded
  __bf16* act_bf = (__bf16*)carve((size_t)MP * KP_IN * 2);      // features/h0/h2 (padded)
  __bf16* h1_bf  = (__bf16*)carve((size_t)MP * KP_H * 2);       // h1 (rows>=NODES unused)
  float*  buf_x  = (float*)carve((size_t)NODES * F_IN * 4);     // x, later h2pre
  float*  hproj1 = (float*)carve((size_t)NODES * HEADS * F_HID * 4);
  float*  gat1   = (float*)carve((size_t)NODES * HEADS * F_HID * 4);
  float*  hproj2 = (float*)carve((size_t)NODES * HEADS * F_OUT * 4);
  float*  gat2   = (float*)carve((size_t)NODES * HEADS * F_OUT * 4);
  float*  bsum   = (float*)carve((size_t)F_IN * 4);
  float*  bsq    = (float*)carve((size_t)F_IN * 4);
  float*  bmean  = (float*)carve((size_t)F_IN * 4);
  float*  bvar   = (float*)carve((size_t)F_IN * 4);
  float*  al1s   = (float*)carve((size_t)NODES * HEADS * 4);
  float*  al1d   = (float*)carve((size_t)NODES * HEADS * 4);
  float*  m1     = (float*)carve((size_t)NODES * HEADS * 4);
  float*  den1   = (float*)carve((size_t)NODES * HEADS * 4);
  float*  e1     = (float*)carve((size_t)EDGES * HEADS * 4);
  float*  al2s   = (float*)carve((size_t)NODES * HEADS * 4);
  float*  al2d   = (float*)carve((size_t)NODES * HEADS * 4);
  float*  m2     = (float*)carve((size_t)NODES * HEADS * 4);
  float*  den2   = (float*)carve((size_t)NODES * HEADS * 4);
  float*  e2     = (float*)carve((size_t)EDGES * HEADS * 4);
  float*  dsum   = (float*)carve((size_t)NDOMS * F_IN * 4);
  float*  dsq    = (float*)carve((size_t)NDOMS * F_IN * 4);
  int*    dcnt   = (int*)carve((size_t)NDOMS * 4);

  const float NEG_INF = -__builtin_huge_valf();
  const int ROW_CHUNKS = 32;
  const int CHUNK = (NODES + ROW_CHUNKS - 1) / ROW_CHUNKS;

  // ---- 1. pack / transpose weights & activations to padded bf16 ----
  cast_pad_kernel<<<grid1((size_t)MP * KP_IN, 256), 256, 0, stream>>>(
      features, act_bf, NODES, F_IN, MP, KP_IN);
  transpose_pad_kernel<<<grid1((size_t)NP_IN * KP_IN, 256), 256, 0, stream>>>(
      fc1_w, wb_fc1, F_IN, F_IN, NP_IN, KP_IN);
  transpose_pad_kernel<<<grid1((size_t)NP_H1 * KP_IN, 256), 256, 0, stream>>>(
      W1, wb_w1, F_IN, HEADS * F_HID, NP_H1, KP_IN);
  transpose_pad_kernel<<<grid1((size_t)NP_O * KP_H, 256), 256, 0, stream>>>(
      W2, wb_w2, F_HID, HEADS * F_OUT, NP_O, KP_H);
  transpose_pad_kernel<<<grid1((size_t)NP_IN * KP_IN, 256), 256, 0, stream>>>(
      fc3_w, wb_fc3, F_IN, F_IN, NP_IN, KP_IN);

  // ---- 2. x = features @ fc1_w + fc1_b ----
  dim3 gb(256);
  dim3 g_fc(NP_IN / 128, MP / 128);
  gemm_bf16_wmma<<<g_fc, gb, 0, stream>>>(act_bf, wb_fc1, buf_x, fc1_b,
                                          NODES, F_IN, F_IN, KP_IN, KP_IN);

  // ---- 3. batchnorm + ELU -> h0 (padded bf16, overwrites act_bf) ----
  hipMemsetAsync(bsum, 0, (size_t)F_IN * 4, stream);
  hipMemsetAsync(bsq, 0, (size_t)F_IN * 4, stream);
  dim3 gcs((F_IN + 255) / 256, ROW_CHUNKS);
  colsum_partial<<<gcs, 256, 0, stream>>>(buf_x, bsum, bsq, NODES, F_IN, CHUNK);
  bn_finalize<<<(F_IN + 255) / 256, 256, 0, stream>>>(bsum, bsq, bmean, bvar, NODES, F_IN);
  bn_apply_elu_kernel<<<grid1((size_t)MP * KP_IN, 256), 256, 0, stream>>>(
      buf_x, bmean, bvar, bn_g, bn_b, act_bf, NODES, F_IN, MP, KP_IN);

  // ---- 4. GAT layer 1 ----
  dim3 g_w1(NP_H1 / 128, MP / 128);
  gemm_bf16_wmma<<<g_w1, gb, 0, stream>>>(act_bf, wb_w1, hproj1, nullptr,
                                          NODES, HEADS * F_HID, F_IN, KP_IN, KP_IN);
  gat_logits<<<grid1((size_t)NODES * HEADS, 256), 256, 0, stream>>>(hproj1, a1_s, a1_d, al1s, al1d, NODES, F_HID);
  fill_f32_kernel<<<grid1((size_t)NODES * HEADS, 256), 256, 0, stream>>>(m1, NEG_INF, (size_t)NODES * HEADS);
  hipMemsetAsync(den1, 0, (size_t)NODES * HEADS * 4, stream);
  hipMemsetAsync(gat1, 0, (size_t)NODES * HEADS * F_HID * 4, stream);
  edge_max_kernel<<<grid1((size_t)EDGES * HEADS, 256), 256, 0, stream>>>(src, dst, al1s, al1d, e1, m1);
  edge_exp_sum_kernel<<<grid1((size_t)EDGES * HEADS, 256), 256, 0, stream>>>(dst, e1, m1, den1);
  scatter_msg_kernel<<<grid1((size_t)EDGES * HEADS * F_HID, 256), 256, 0, stream>>>(
      src, dst, e1, den1, hproj1, gat1, F_HID);
  head_mean_kernel<<<grid1((size_t)NODES * F_HID, 256), 256, 0, stream>>>(
      gat1, nullptr, h1_bf, NODES, F_HID, 1);

  // ---- 5. GAT layer 2 -> z ----
  dim3 g_w2(NP_O / 128, MP / 128);
  gemm_bf16_wmma<<<g_w2, gb, 0, stream>>>(h1_bf, wb_w2, hproj2, nullptr,
                                          NODES, HEADS * F_OUT, F_HID, KP_H, KP_H);
  gat_logits<<<grid1((size_t)NODES * HEADS, 256), 256, 0, stream>>>(hproj2, a2_s, a2_d, al2s, al2d, NODES, F_OUT);
  fill_f32_kernel<<<grid1((size_t)NODES * HEADS, 256), 256, 0, stream>>>(m2, NEG_INF, (size_t)NODES * HEADS);
  hipMemsetAsync(den2, 0, (size_t)NODES * HEADS * 4, stream);
  hipMemsetAsync(gat2, 0, (size_t)NODES * HEADS * F_OUT * 4, stream);
  edge_max_kernel<<<grid1((size_t)EDGES * HEADS, 256), 256, 0, stream>>>(src, dst, al2s, al2d, e2, m2);
  edge_exp_sum_kernel<<<grid1((size_t)EDGES * HEADS, 256), 256, 0, stream>>>(dst, e2, m2, den2);
  scatter_msg_kernel<<<grid1((size_t)EDGES * HEADS * F_OUT, 256), 256, 0, stream>>>(
      src, dst, e2, den2, hproj2, gat2, F_OUT);
  head_mean_kernel<<<grid1((size_t)NODES * F_OUT, 256), 256, 0, stream>>>(
      gat2, out_z, nullptr, NODES, F_OUT, 0);

  // ---- 6. MLP head -> y_pred ----
  mlp_kernel<<<grid1((size_t)NODES, 256), 256, 0, stream>>>(out_z, mw1, mb1, mw2, mb2, mlp_flag, out_yp);

  // ---- 7. fc2 -> h2pre (reuse buf_x) ----
  fc2_kernel<<<grid1((size_t)NODES * F_IN, 256), 256, 0, stream>>>(out_z, fc2_w, fc2_b, buf_x);

  // ---- 8. DSBN + ELU -> h2 (padded bf16, overwrites act_bf) ----
  hipMemsetAsync(dsum, 0, (size_t)NDOMS * F_IN * 4, stream);
  hipMemsetAsync(dsq, 0, (size_t)NDOMS * F_IN * 4, stream);
  hipMemsetAsync(dcnt, 0, (size_t)NDOMS * 4, stream);
  dsbn_count_kernel<<<grid1((size_t)NODES, 256), 256, 0, stream>>>(y, dcnt);
  dsbn_partial<<<gcs, 256, 0, stream>>>(buf_x, y, dsum, dsq, NODES, F_IN, CHUNK);
  dsbn_apply_kernel<<<grid1((size_t)MP * KP_IN, 256), 256, 0, stream>>>(
      buf_x, y, dcnt, dsum, dsq, dg, db, act_bf, NODES, F_IN, MP, KP_IN);

  // ---- 9. h3 = h2 @ fc3_w + fc3_b -> d_out ----
  gemm_bf16_wmma<<<g_fc, gb, 0, stream>>>(act_bf, wb_fc3, out_h3, fc3_b,
                                          NODES, F_IN, F_IN, KP_IN, KP_IN);
}
